// MinkowskiSafeDeconv_15479062134887
// MI455X (gfx1250) — matrix operand order repaired and checked by reading, not compiled, not run on hardware
//
#include <hip/hip_runtime.h>
#include <hip/hip_bf16.h>

typedef __attribute__((ext_vector_type(16))) __bf16 v16bf;
typedef __attribute__((ext_vector_type(8)))  float  v8f;

#define NIN   64
#define NOUT  32
#define KOFF  9
#define FRAG_ELEMS 512                    // bf16 elems per 32x16 B fragment
#define NFRAG (KOFF * 4)                  // 2 K-steps x 2 N-col blocks per offset
#define WPACK_ELEMS (NFRAG * FRAG_ELEMS)  // 18432 bf16 = 36 KB

// ---------------------------------------------------------------------------
// Convert fp32 weights [K][NIN][NOUT] -> bf16, pre-swizzled into the wave32
// WMMA B-fragment layout, contiguous k order (same as A): for fragment f = k*4 + ncol*2 + kstep,
// lane L holds column n = ncol*16 + (L&15), K-values kstep*32 + (L>>4)*16 + j
// for j = 0..15, stored contiguously so the GEMM does one 32B load per lane.
// ---------------------------------------------------------------------------
__global__ void pack_weights_kernel(const float* __restrict__ w,
                                    __bf16* __restrict__ wp) {
  for (int i = threadIdx.x; i < WPACK_ELEMS; i += blockDim.x) {
    int f      = i >> 9;          // /512
    int within = i & 511;
    int lane   = within >> 4;
    int j      = within & 15;
    int k      = f >> 2;
    int ncol   = (f >> 1) & 1;
    int kstep  = f & 1;
    int n  = ncol * 16 + (lane & 15);
    int kk = kstep * 32 + ((lane >> 4) << 4) + j;
    wp[i] = (__bf16)w[(k * NIN + kk) * NOUT + n];
  }
}

// ---------------------------------------------------------------------------
// Each wave owns a 16-row tile of in_feats. A-fragments (16x32 bf16) follow the
// contiguous k order (as in B): lane L holds row M = L&15; lanes 0-15 hold K
// [base..base+15] with base = 0, lanes 16-31 with base = 16 (+32 for K-step 1).
// 4 WMMAs per kernel offset (2 K-steps x 2 N-col blocks), then scatter the
// 16x32 fp32 C tile with global fp32 atomics using out_idx. Index loads are
// batched ahead of the atomic phase so they issue as one clause (one wait),
// maximizing memory-level parallelism on the scatter-bound path.
// ---------------------------------------------------------------------------
__global__ void __launch_bounds__(256)
deconv_gemm_scatter_kernel(const float* __restrict__ in_feats,
                           const __bf16* __restrict__ wpack,
                           const int* __restrict__ out_idx,
                           float* __restrict__ out,
                           int n_rows) {
  __shared__ __align__(16) __bf16 ldsW[WPACK_ELEMS];

  // Stage the 36 KB packed-weight image into LDS (block-cooperative b128 copies).
  {
    const uint4* src = (const uint4*)wpack;
    uint4*       dst = (uint4*)ldsW;
    const int n16 = (WPACK_ELEMS * 2) / 16;   // 2304 x 16B
    for (int i = threadIdx.x; i < n16; i += blockDim.x) dst[i] = src[i];
  }
  __syncthreads();

  const int wave = threadIdx.x >> 5;
  const int lane = threadIdx.x & 31;
  const int tile = blockIdx.x * 8 + wave;
  const int row_base = tile * 16;
  if (row_base >= n_rows) return;   // wave-uniform; EXEC stays all-1s for WMMA

  const int m  = lane & 15;
  const int hi = lane >> 4;
  const float* arow = in_feats + (size_t)(row_base + m) * NIN;

  // Speculative prefetch of the next block's A rows (global_prefetch_b8).
  if (row_base + 8 * 16 < n_rows)
    __builtin_prefetch(arow + 8 * 16 * NIN, 0, 0);

  // Build both A fragments (K = 0..31 and 32..63), fp32 -> bf16 on the fly.
  v16bf a[2];
#pragma unroll
  for (int s = 0; s < 2; ++s) {
    const int base = s * 32 + hi * 16;
#pragma unroll
    for (int j = 0; j < 8; ++j) {
      a[s][j]     = (__bf16)arow[base + j];
      a[s][j + 8] = (__bf16)arow[base + 8 + j];
    }
  }

  const v16bf* bf = (const v16bf*)ldsW;   // fragment f: bf[f*32 + lane]
  // Per-lane base into out_idx for this wave's 8 C rows (M = hi*8 + r).
  const int* idx_base = out_idx + (size_t)(row_base + hi * 8) * KOFF;

#pragma unroll 1
  for (int k = 0; k < KOFF; ++k) {
    v16bf b00 = bf[(k * 4 + 0) * 32 + lane];   // ncol 0, kstep 0
    v16bf b01 = bf[(k * 4 + 1) * 32 + lane];   // ncol 0, kstep 1
    v16bf b10 = bf[(k * 4 + 2) * 32 + lane];   // ncol 1, kstep 0
    v16bf b11 = bf[(k * 4 + 3) * 32 + lane];   // ncol 1, kstep 1

    // Phase 1: batch the 8 output-row index loads (single clause, one wait).
    int gidx[8];
#pragma unroll
    for (int r = 0; r < 8; ++r)
      gidx[r] = idx_base[r * KOFF + k];

    v8f c0 = {};
    v8f c1 = {};
    c0 = __builtin_amdgcn_wmma_f32_16x16x32_bf16(false, a[0], false, b00,
                                                 (short)0, c0, false, false);
    c0 = __builtin_amdgcn_wmma_f32_16x16x32_bf16(false, a[1], false, b01,
                                                 (short)0, c0, false, false);
    c1 = __builtin_amdgcn_wmma_f32_16x16x32_bf16(false, a[0], false, b10,
                                                 (short)0, c1, false, false);
    c1 = __builtin_amdgcn_wmma_f32_16x16x32_bf16(false, a[1], false, b11,
                                                 (short)0, c1, false, false);

    // Phase 2: scatter. C-fragment VGPR r holds row M = hi*8 + r, column
    // N = lane&15 (c0 = cols 0..15, c1 = cols 16..31). Native fp32 atomics.
#pragma unroll
    for (int r = 0; r < 8; ++r) {
      float* dst = out + (size_t)gidx[r] * NOUT + (lane & 15);
      unsafeAtomicAdd(dst,      c0[r]);
      unsafeAtomicAdd(dst + 16, c1[r]);
    }
  }
}

__global__ void bias_relu_kernel(float* __restrict__ out,
                                 const float* __restrict__ bias,
                                 int total) {
  int i = blockIdx.x * blockDim.x + threadIdx.x;
  if (i < total) {
    float v = out[i] + bias[i & (NOUT - 1)];
    out[i] = v > 0.0f ? v : 0.0f;
  }
}

extern "C" void kernel_launch(void* const* d_in, const int* in_sizes, int n_in,
                              void* d_out, int out_size, void* d_ws, size_t ws_size,
                              hipStream_t stream) {
  const float* in_feats = (const float*)d_in[0];
  const float* weight   = (const float*)d_in[1];
  const float* bias     = (const float*)d_in[2];
  const int*   out_idx  = (const int*)d_in[3];
  // d_in[4] = num_out scalar; recover it from out_size instead.
  float*  out = (float*)d_out;
  __bf16* wp  = (__bf16*)d_ws;

  const int n_rows = in_sizes[0] / NIN;     // 262144

  // Output is accumulated with atomics: zero it every call (graph-capturable).
  hipMemsetAsync(d_out, 0, (size_t)out_size * sizeof(float), stream);

  pack_weights_kernel<<<1, 256, 0, stream>>>(weight, wp);

  const int tiles  = n_rows / 16;           // 16 rows per wave
  const int blocks = (tiles + 7) / 8;       // 8 waves per block
  deconv_gemm_scatter_kernel<<<blocks, 256, 0, stream>>>(in_feats, wp, out_idx,
                                                         out, n_rows);

  bias_relu_kernel<<<(out_size + 255) / 256, 256, 0, stream>>>(out, bias,
                                                               out_size);
}
